// Matching_86242943304240
// MI455X (gfx1250) — matrix operand order repaired and checked by reading, not compiled, not run on hardware
//
#include <hip/hip_runtime.h>

typedef __attribute__((ext_vector_type(16))) __bf16 v16bf;
typedef __attribute__((ext_vector_type(8)))  float  v8f;

#define NBATCH 32
#define CCH    256
#define WH     4096
#define SCL    0.0625f   // 1/sqrt(256)

__device__ __forceinline__ unsigned short f32_to_bf16(float f) {
  unsigned u = __float_as_uint(f);
  u += 0x7FFFu + ((u >> 16) & 1u);           // round-to-nearest-even
  return (unsigned short)(u >> 16);
}
__device__ __forceinline__ float bf16_to_f32(unsigned short h) {
  return __uint_as_float(((unsigned)h) << 16);
}

// ---------------- 1. q = leaky_relu(style @ w_dense, 0.3) ----------------
__global__ void q_kernel(const float* __restrict__ style,
                         const float* __restrict__ w_dense,
                         float* __restrict__ q) {
  __shared__ float s_style[CCH];
  int n = blockIdx.x, c = threadIdx.x;
  s_style[c] = style[n * CCH + c];
  __syncthreads();
  float acc = 0.f;
#pragma unroll 4
  for (int d = 0; d < CCH; ++d) acc += s_style[d] * w_dense[d * CCH + c];
  q[n * CCH + c] = acc > 0.f ? acc : 0.3f * acc;
}

// -------- 2. one pass over inputs: pa_logits + column-sum partials --------
// grid (n=32, ptile=16), block 256 (8 waves). Wave w handles 32 positions;
// lane l owns channels 8l..8l+7 (two float4 loads -> fully coalesced rows).
__global__ void stats_kernel(const float* __restrict__ x,
                             const float* __restrict__ q,
                             float* __restrict__ pa_logits,
                             float* __restrict__ S_part) {
  int n = blockIdx.x;
  int ptile = blockIdx.y;
  int wave = threadIdx.x >> 5, lane = threadIdx.x & 31;
  __shared__ float s_ws[8 * CCH];

  const float* xb = x + (size_t)n * WH * CCH;
  float q0[8];
#pragma unroll
  for (int i = 0; i < 8; ++i) q0[i] = q[n * CCH + lane * 8 + i];
  float colsum[8];
#pragma unroll
  for (int i = 0; i < 8; ++i) colsum[i] = 0.f;

  int p0 = ptile * 256 + wave * 32;
  for (int t = 0; t < 32; ++t) {
    int p = p0 + t;
    const float4* row = (const float4*)(xb + (size_t)p * CCH + lane * 8);
    float4 a = row[0], b = row[1];
    float dot = a.x * q0[0] + a.y * q0[1] + a.z * q0[2] + a.w * q0[3] +
                b.x * q0[4] + b.y * q0[5] + b.z * q0[6] + b.w * q0[7];
    colsum[0] += a.x; colsum[1] += a.y; colsum[2] += a.z; colsum[3] += a.w;
    colsum[4] += b.x; colsum[5] += b.y; colsum[6] += b.z; colsum[7] += b.w;
#pragma unroll
    for (int off = 16; off > 0; off >>= 1) dot += __shfl_xor(dot, off, 32);
    if (lane == 0) pa_logits[n * WH + p] = dot * SCL;
  }
  // deterministic per-block column reduction (no float atomics)
#pragma unroll
  for (int i = 0; i < 8; ++i) s_ws[wave * CCH + lane * 8 + i] = colsum[i];
  __syncthreads();
  int c = threadIdx.x;
  float tot = 0.f;
#pragma unroll
  for (int w = 0; w < 8; ++w) tot += s_ws[w * CCH + c];
  S_part[(n * 16 + ptile) * CCH + c] = tot;
}

// ------------- 3. ca = softmax_c(q * colsum * scale) ----------------------
__global__ void ca_kernel(const float* __restrict__ S_part,
                          const float* __restrict__ q,
                          float* __restrict__ ca) {
  int n = blockIdx.x, c = threadIdx.x;
  float s = 0.f;
#pragma unroll
  for (int t = 0; t < 16; ++t) s += S_part[(n * 16 + t) * CCH + c];
  float l = s * q[n * CCH + c] * SCL;
  __shared__ float red[CCH];
  red[c] = l; __syncthreads();
  for (int st = 128; st > 0; st >>= 1) { if (c < st) red[c] = fmaxf(red[c], red[c + st]); __syncthreads(); }
  float m = red[0]; __syncthreads();
  float e = __expf(l - m);
  red[c] = e; __syncthreads();
  for (int st = 128; st > 0; st >>= 1) { if (c < st) red[c] += red[c + st]; __syncthreads(); }
  ca[n * CCH + c] = e / red[0];
}

// ------------- 4. pa = softmax over 4096 positions, in place --------------
__global__ void pa_kernel(float* __restrict__ pa) {
  int n = blockIdx.x, t = threadIdx.x;   // block 1024
  float v[4], m = -1e30f;
  __shared__ float red[1024];
#pragma unroll
  for (int i = 0; i < 4; ++i) { v[i] = pa[n * WH + t + i * 1024]; m = fmaxf(m, v[i]); }
  red[t] = m; __syncthreads();
  for (int st = 512; st > 0; st >>= 1) { if (t < st) red[t] = fmaxf(red[t], red[t + st]); __syncthreads(); }
  m = red[0]; __syncthreads();
  float s = 0.f;
#pragma unroll
  for (int i = 0; i < 4; ++i) { v[i] = __expf(v[i] - m); s += v[i]; }
  red[t] = s; __syncthreads();
  for (int st = 512; st > 0; st >>= 1) { if (t < st) red[t] += red[t + st]; __syncthreads(); }
  float inv = 1.f / red[0];
#pragma unroll
  for (int i = 0; i < 4; ++i) pa[n * WH + t + i * 1024] = v[i] * inv;
}

// ---- 5. Wn^T[n][co][ci] = ca[n,ci]*wconv[ci,co] as bf16 hi/lo planes -----
__global__ void prep_w(const float* __restrict__ wconv,
                       const float* __restrict__ ca,
                       unsigned short* __restrict__ wT_hi,
                       unsigned short* __restrict__ wT_lo) {
  int idx = blockIdx.x * blockDim.x + threadIdx.x;  // (n*256+co)*256+ci
  int ci = idx & 255;
  int co = (idx >> 8) & 255;
  int n  = idx >> 16;
  float w = ca[n * CCH + ci] * wconv[ci * CCH + co];
  unsigned short hi = f32_to_bf16(w);
  float lo = w - bf16_to_f32(hi);
  wT_hi[idx] = hi;
  wT_lo[idx] = f32_to_bf16(lo);
}

// ------------- 6. batched GEMM with split-bf16 x3 WMMA --------------------
#define MB 128
#define NBLK 64
#define KC 64
#define LDA 72
#define LDB 72

union FragU { v16bf v; uint4 u[2]; };

__global__ __launch_bounds__(256) void gemm_kernel(
    const float* __restrict__ x,
    const unsigned short* __restrict__ wT_hi,
    const unsigned short* __restrict__ wT_lo,
    const float* __restrict__ pa,
    float* __restrict__ out) {
  __shared__ unsigned short sAh[MB * LDA];
  __shared__ unsigned short sAl[MB * LDA];
  __shared__ unsigned short sBh[NBLK * LDB];
  __shared__ unsigned short sBl[NBLK * LDB];
  __shared__ float sPa[MB];

  int co_base = blockIdx.x * NBLK;
  int m_base  = blockIdx.y * MB;
  int n       = blockIdx.z;
  int tid = threadIdx.x;
  int wave = tid >> 5, lane = tid & 31;
  int wm = (wave & 3) * 32;   // wave row offset   (4 waves along M)
  int wn = (wave >> 2) * 32;  // wave col offset   (2 waves along N)

  if (tid < MB) sPa[tid] = pa[n * WH + m_base + tid];

  const float* xb = x + ((size_t)n * WH + m_base) * CCH;
  const unsigned short* wh = wT_hi + ((size_t)n * CCH + co_base) * CCH;
  const unsigned short* wl = wT_lo + ((size_t)n * CCH + co_base) * CCH;

  v8f acc[2][2] = {};

  for (int kc = 0; kc < CCH; kc += KC) {
    __syncthreads();
    // stage B planes via CDNA5 async global->LDS DMA (bypasses VGPRs,
    // co-executes with the VALU bf16-split work below). 16B per lane,
    // 2 iterations x 2 planes cover 64(co) x 64(ci) bf16 each.
#pragma unroll
    for (int it = 0; it < 2; ++it) {
      int idx = it * 256 + tid;          // b128 chunk index, 8 per row
      int row = idx >> 3;
      int c8  = (idx & 7) * 8;
      unsigned goff = (unsigned)(((size_t)row * CCH + kc + c8) * 2);
      unsigned ldst_h = (unsigned)(uintptr_t)&sBh[row * LDB + c8];
      unsigned ldst_l = (unsigned)(uintptr_t)&sBl[row * LDB + c8];
      asm volatile("global_load_async_to_lds_b128 %0, %1, %2"
                   :: "v"(ldst_h), "v"(goff), "s"(wh) : "memory");
      asm volatile("global_load_async_to_lds_b128 %0, %1, %2"
                   :: "v"(ldst_l), "v"(goff), "s"(wl) : "memory");
    }
    // stage A: 128 rows x 64 f32 -> bf16 hi/lo   (8 float4 iters/thread)
#pragma unroll
    for (int it = 0; it < 8; ++it) {
      int idx = it * 256 + tid;            // float4 index, 16 per row
      int row = idx >> 4;
      int c4  = (idx & 15) * 4;
      float4 f = *(const float4*)(xb + (size_t)row * CCH + kc + c4);
      unsigned short h0 = f32_to_bf16(f.x), h1 = f32_to_bf16(f.y),
                     h2 = f32_to_bf16(f.z), h3 = f32_to_bf16(f.w);
      unsigned short l0 = f32_to_bf16(f.x - bf16_to_f32(h0));
      unsigned short l1 = f32_to_bf16(f.y - bf16_to_f32(h1));
      unsigned short l2 = f32_to_bf16(f.z - bf16_to_f32(h2));
      unsigned short l3 = f32_to_bf16(f.w - bf16_to_f32(h3));
      uint2 ph; ph.x = (unsigned)h0 | ((unsigned)h1 << 16); ph.y = (unsigned)h2 | ((unsigned)h3 << 16);
      uint2 pl; pl.x = (unsigned)l0 | ((unsigned)l1 << 16); pl.y = (unsigned)l2 | ((unsigned)l3 << 16);
      *(uint2*)&sAh[row * LDA + c4] = ph;
      *(uint2*)&sAl[row * LDA + c4] = pl;
    }
    asm volatile("s_wait_asynccnt 0x0" ::: "memory");
    __syncthreads();

#pragma unroll
    for (int k0 = 0; k0 < KC; k0 += 32) {
      FragU ah[2], al[2], bh[2], bl[2];
      // A fragment: lanes 0-15 rows M, K chunks {k0+0..7, k0+16..23};
      //             lanes 16-31 same rows, chunks {k0+8..15, k0+24..31}
      int arow = wm + (lane & 15);
      int koff = k0 + ((lane >> 4) << 3);
#pragma unroll
      for (int i = 0; i < 2; ++i) {
        const unsigned short* pA = &sAh[(arow + i * 16) * LDA + koff];
        ah[i].u[0] = *(const uint4*)pA;
        ah[i].u[1] = *(const uint4*)(pA + 16);
        const unsigned short* pL = &sAl[(arow + i * 16) * LDA + koff];
        al[i].u[0] = *(const uint4*)pL;
        al[i].u[1] = *(const uint4*)(pL + 16);
      }
      // B fragment: lane = column co; lanes 0-15 K chunks {k0+0..7, k0+16..23}, 16-31 {k0+8..15, k0+24..31}
      int bco = wn + (lane & 15);
      int bk  = k0 + ((lane >> 4) << 3);
#pragma unroll
      for (int j = 0; j < 2; ++j) {
        const unsigned short* pB = &sBh[(bco + j * 16) * LDB + bk];
        bh[j].u[0] = *(const uint4*)pB;
        bh[j].u[1] = *(const uint4*)(pB + 16);
        const unsigned short* pM = &sBl[(bco + j * 16) * LDB + bk];
        bl[j].u[0] = *(const uint4*)pM;
        bl[j].u[1] = *(const uint4*)(pM + 16);
      }
#pragma unroll
      for (int i = 0; i < 2; ++i)
#pragma unroll
        for (int j = 0; j < 2; ++j) {
          acc[i][j] = __builtin_amdgcn_wmma_f32_16x16x32_bf16(
              false, ah[i].v, false, bh[j].v, (short)0, acc[i][j], false, false);
          acc[i][j] = __builtin_amdgcn_wmma_f32_16x16x32_bf16(
              false, al[i].v, false, bh[j].v, (short)0, acc[i][j], false, false);
          acc[i][j] = __builtin_amdgcn_wmma_f32_16x16x32_bf16(
              false, ah[i].v, false, bl[j].v, (short)0, acc[i][j], false, false);
        }
    }
  }

  // epilogue: row-scale by pa, leaky_relu(0.1), store
  float* ob = out + ((size_t)n * WH + m_base) * CCH + co_base;
#pragma unroll
  for (int i = 0; i < 2; ++i)
#pragma unroll
    for (int j = 0; j < 2; ++j)
#pragma unroll
      for (int r = 0; r < 8; ++r) {
        int row = wm + i * 16 + ((lane >> 4) << 3) + r;
        int col = wn + j * 16 + (lane & 15);
        float v = acc[i][j][r] * sPa[row];
        v = v > 0.f ? v : 0.1f * v;
        ob[(size_t)row * CCH + col] = v;
      }
}

extern "C" void kernel_launch(void* const* d_in, const int* in_sizes, int n_in,
                              void* d_out, int out_size, void* d_ws, size_t ws_size,
                              hipStream_t stream) {
  const float* x       = (const float*)d_in[0];  // [32,64,64,256]
  const float* style   = (const float*)d_in[1];  // [32,256]
  const float* w_dense = (const float*)d_in[2];  // [256,256]
  const float* w_conv  = (const float*)d_in[3];  // [1,1,256,256]
  float* out = (float*)d_out;

  char* ws = (char*)d_ws;
  size_t off = 0;
  float* q  = (float*)(ws + off); off += (size_t)NBATCH * CCH * 4;        // 32 KB
  float* ca = (float*)(ws + off); off += (size_t)NBATCH * CCH * 4;        // 32 KB
  float* Sp = (float*)(ws + off); off += (size_t)NBATCH * 16 * CCH * 4;   // 512 KB
  float* pa = (float*)(ws + off); off += (size_t)NBATCH * WH * 4;         // 512 KB
  unsigned short* wTh = (unsigned short*)(ws + off); off += (size_t)NBATCH * CCH * CCH * 2; // 4 MB
  unsigned short* wTl = (unsigned short*)(ws + off); off += (size_t)NBATCH * CCH * CCH * 2; // 4 MB

  q_kernel<<<NBATCH, 256, 0, stream>>>(style, w_dense, q);
  stats_kernel<<<dim3(NBATCH, 16), 256, 0, stream>>>(x, q, pa, Sp);
  ca_kernel<<<NBATCH, 256, 0, stream>>>(Sp, q, ca);
  pa_kernel<<<NBATCH, 1024, 0, stream>>>(pa);
  prep_w<<<(NBATCH * CCH * CCH) / 256, 256, 0, stream>>>(w_conv, ca, wTh, wTl);
  gemm_kernel<<<dim3(CCH / NBLK, WH / MB, NBATCH), 256, 0, stream>>>(x, wTh, wTl, pa, out);
}
